// NRUCell_1039382085932
// MI455X (gfx1250) — matrix operand order repaired and checked by reading, not compiled, not run on hardware
//
#include <hip/hip_runtime.h>
#include <math.h>

// ---------- types ----------
typedef __attribute__((ext_vector_type(16))) __bf16 v16bf;
typedef __attribute__((ext_vector_type(8)))  __bf16 v8bf;
typedef __attribute__((ext_vector_type(8)))  float  v8f;

#define BATCH 8192
#define INPF  1024
#define HIDF  1024
#define MEMF  256
#define KTOT  2304   // INP + HID + MEM
#define K2    1280   // HID + MEM
#define N2PAD 144    // 136 real cols padded to 9*16
#define LDSW  40     // 32 + 8 bf16 pad -> 80B row stride, 16B aligned, conflict-free

// ---------- WMMA wrapper (CDNA5 bf16, f32 accum) ----------
__device__ __forceinline__ v8f wmma_bf16(v16bf a, v16bf b, v8f c) {
  // (neg_a, A, neg_b, B, c_mod, C, reuse_a, reuse_b)
  return __builtin_amdgcn_wmma_f32_16x16x32_bf16(false, a, false, b, (short)0, c,
                                                 false, false);
}

// split fp32 -> bf16 hi + bf16 lo residual (bf16x2 split precision)
__device__ __forceinline__ void cvt_store4(__bf16* ph, __bf16* pl, float4 v) {
  float f0 = v.x, f1 = v.y, f2 = v.z, f3 = v.w;
  __bf16 h0 = (__bf16)f0, h1 = (__bf16)f1, h2 = (__bf16)f2, h3 = (__bf16)f3;
  ph[0] = h0; ph[1] = h1; ph[2] = h2; ph[3] = h3;
  pl[0] = (__bf16)(f0 - (float)h0);
  pl[1] = (__bf16)(f1 - (float)h1);
  pl[2] = (__bf16)(f2 - (float)h2);
  pl[3] = (__bf16)(f3 - (float)h3);
}

// A-matrix 16x32 bf16 frag (ISA 7.12.2): lane m<16: K 0-7 & 16-23; lane m+16: K 8-15 & 24-31
__device__ __forceinline__ v16bf load_frag_a(const __bf16* tile, int rowBase, int lane) {
  int r  = rowBase + (lane & 15);
  int k0 = (lane >> 4) * 8;
  const __bf16* p = tile + r * LDSW;
  v8bf c0 = *(const v8bf*)(p + k0);        // ds_load_b128
  v8bf c1 = *(const v8bf*)(p + k0 + 16);   // ds_load_b128
  v16bf out;
#pragma unroll
  for (int i = 0; i < 8; ++i) { out[i] = c0[i]; out[i + 8] = c1[i]; }
  return out;
}

// B-matrix 32x16 bf16 frag: lane n<16 holds column n, K 0-7 & 16-23; lane n+16: K 8-15 & 24-31
// (tile stored as W rows x K, i.e. B columns x K)
__device__ __forceinline__ v16bf load_frag_b(const __bf16* tile, int colBase, int lane) {
  int c  = colBase + (lane & 15);
  int k0 = (lane >> 4) * 8;
  const __bf16* p = tile + c * LDSW;
  v8bf c0 = *(const v8bf*)(p + k0);
  v8bf c1 = *(const v8bf*)(p + k0 + 16);
  v16bf out;
#pragma unroll
  for (int i = 0; i < 8; ++i) { out[i] = c0[i]; out[i + 8] = c1[i]; }
  return out;
}

// ============================================================================
// GEMM1: h = relu([x|h0|mem] (8192x2304) @ W_h^T (2304x1024) + b_h)
// block = 128x128 tile, 256 threads = 8 waves (4m x 2n), wave tile 32x64, BK=32
// 24 wmma per wave per K-step vs 24 ds_b128 frag loads: 1:1; conversions hidden
// under XDL/TRANS co-execution.
// ============================================================================
__global__ __launch_bounds__(256) void nru_gemm1(
    const float* __restrict__ x, const float* __restrict__ h0,
    const float* __restrict__ memin, const float* __restrict__ Wh,
    const float* __restrict__ bh, float* __restrict__ hOut) {
  __shared__ __align__(16) __bf16 Ah[128 * LDSW];
  __shared__ __align__(16) __bf16 Al[128 * LDSW];
  __shared__ __align__(16) __bf16 Bh[128 * LDSW];
  __shared__ __align__(16) __bf16 Bl[128 * LDSW];

  const int tid   = threadIdx.x;
  const int lane  = tid & 31;
  const int wave  = tid >> 5;
  const int wm    = (wave & 3) * 32;   // 4 wave-rows
  const int wn    = (wave >> 2) * 64;  // 2 wave-cols
  const int mBase = blockIdx.y * 128;
  const int nBase = blockIdx.x * 128;

  v8f acc[2][4];
#pragma unroll
  for (int i = 0; i < 2; ++i)
#pragma unroll
    for (int j = 0; j < 4; ++j)
#pragma unroll
      for (int e = 0; e < 8; ++e) acc[i][j][e] = 0.f;

  for (int kb = 0; kb < KTOT; kb += 32) {
    // gather concat source (boundaries are multiples of 32)
    const float* src; int col, ldk;
    if (kb < INPF)             { src = x;     col = kb;               ldk = INPF; }
    else if (kb < INPF + HIDF) { src = h0;    col = kb - INPF;        ldk = HIDF; }
    else                       { src = memin; col = kb - INPF - HIDF; ldk = MEMF; }

    // A tile 128x32 fp32 -> bf16 hi/lo (4 float4 per thread)
#pragma unroll
    for (int i = 0; i < 4; ++i) {
      int s = tid + 256 * i;
      int row = s >> 3;
      int c4  = (s & 7) << 2;
      float4 v = *(const float4*)(src + (size_t)(mBase + row) * ldk + col + c4);
      cvt_store4(Ah + row * LDSW + c4, Al + row * LDSW + c4, v);
    }
    // B tile: 128 rows of W_h x 32 K (4 float4 per thread)
#pragma unroll
    for (int i = 0; i < 4; ++i) {
      int s = tid + 256 * i;
      int row = s >> 3;
      int c4  = (s & 7) << 2;
      float4 v = *(const float4*)(Wh + (size_t)(nBase + row) * KTOT + kb + c4);
      cvt_store4(Bh + row * LDSW + c4, Bl + row * LDSW + c4, v);
    }
    __syncthreads();

    v16bf aH[2], aL[2], bH[4], bL[4];
#pragma unroll
    for (int im = 0; im < 2; ++im) {
      aH[im] = load_frag_a(Ah, wm + im * 16, lane);
      aL[im] = load_frag_a(Al, wm + im * 16, lane);
    }
#pragma unroll
    for (int in = 0; in < 4; ++in) {
      bH[in] = load_frag_b(Bh, wn + in * 16, lane);
      bL[in] = load_frag_b(Bl, wn + in * 16, lane);
    }
    // (Ah+Al)(Bh+Bl) ~= Ah*Bh + Ah*Bl + Al*Bh  (fp32-grade precision)
#pragma unroll
    for (int im = 0; im < 2; ++im)
#pragma unroll
      for (int in = 0; in < 4; ++in) {
        acc[im][in] = wmma_bf16(aH[im], bH[in], acc[im][in]);
        acc[im][in] = wmma_bf16(aH[im], bL[in], acc[im][in]);
        acc[im][in] = wmma_bf16(aL[im], bH[in], acc[im][in]);
      }
    __syncthreads();
  }

  // epilogue: bias + relu -> h  (C layout: VGPR r, lanes<16 M=r, lanes>=16 M=r+8)
  const int hi = lane >> 4;
  const int ln = lane & 15;
#pragma unroll
  for (int im = 0; im < 2; ++im)
#pragma unroll
    for (int in = 0; in < 4; ++in) {
      int n = nBase + wn + in * 16 + ln;
      float bias = bh[n];
#pragma unroll
      for (int r = 0; r < 8; ++r) {
        int m = mBase + wm + im * 16 + r + 8 * hi;
        hOut[(size_t)m * HIDF + n] = fmaxf(acc[im][in][r] + bias, 0.f);
      }
    }
}

// ============================================================================
// GEMM2: U = [h|mem] (8192x1280) @ W2^T (1280x144pad) + b2
// W2 rows gathered from {W_a, W_b, W_va, W_vb}, zero-padded rows >=136.
// block = 128x16 tile, 8 waves each own one 16x16 C tile.
// ============================================================================
__global__ __launch_bounds__(256) void nru_gemm2(
    const float* __restrict__ hIn, const float* __restrict__ memin,
    const float* __restrict__ Wa, const float* __restrict__ Wb,
    const float* __restrict__ Wva, const float* __restrict__ Wvb,
    const float* __restrict__ ba, const float* __restrict__ bb,
    const float* __restrict__ bva, const float* __restrict__ bvb,
    float* __restrict__ U) {
  __shared__ __align__(16) __bf16 Ah[128 * LDSW];
  __shared__ __align__(16) __bf16 Al[128 * LDSW];
  __shared__ __align__(16) __bf16 Bh[16 * LDSW];
  __shared__ __align__(16) __bf16 Bl[16 * LDSW];

  const int tid   = threadIdx.x;
  const int lane  = tid & 31;
  const int wave  = tid >> 5;
  const int mBase = blockIdx.y * 128;
  const int nBase = blockIdx.x * 16;

  // per-thread row gather for W2 (threads 0..127: 16 rows x 8 float4)
  const float* wsrc = nullptr; int wrow = 0, brow = 0, bc4 = 0;
  if (tid < 128) {
    brow = tid >> 3; bc4 = (tid & 7) << 2;
    int n = nBase + brow;
    if      (n < 4)   { wsrc = Wa;  wrow = n; }
    else if (n < 8)   { wsrc = Wb;  wrow = n - 4; }
    else if (n < 72)  { wsrc = Wva; wrow = n - 8; }
    else if (n < 136) { wsrc = Wvb; wrow = n - 72; }
  }

  v8f acc;
#pragma unroll
  for (int e = 0; e < 8; ++e) acc[e] = 0.f;

  for (int kb = 0; kb < K2; kb += 32) {
    const float* src; int col, ldk;
    if (kb < HIDF) { src = hIn;   col = kb;        ldk = HIDF; }
    else           { src = memin; col = kb - HIDF; ldk = MEMF; }

#pragma unroll
    for (int i = 0; i < 4; ++i) {
      int s = tid + 256 * i;
      int row = s >> 3;
      int c4  = (s & 7) << 2;
      float4 v = *(const float4*)(src + (size_t)(mBase + row) * ldk + col + c4);
      cvt_store4(Ah + row * LDSW + c4, Al + row * LDSW + c4, v);
    }
    if (tid < 128) {
      float4 v;
      if (wsrc) v = *(const float4*)(wsrc + (size_t)wrow * K2 + kb + bc4);
      else      { v.x = 0.f; v.y = 0.f; v.z = 0.f; v.w = 0.f; }
      cvt_store4(Bh + brow * LDSW + bc4, Bl + brow * LDSW + bc4, v);
    }
    __syncthreads();

    v16bf aH = load_frag_a(Ah, wave * 16, lane);
    v16bf aL = load_frag_a(Al, wave * 16, lane);
    v16bf bH = load_frag_b(Bh, 0, lane);
    v16bf bL = load_frag_b(Bl, 0, lane);
    acc = wmma_bf16(aH, bH, acc);
    acc = wmma_bf16(aH, bL, acc);
    acc = wmma_bf16(aL, bH, acc);
    __syncthreads();
  }

  const int hi = lane >> 4;
  const int ln = lane & 15;
  int n = nBase + ln;
  float bias;
  if      (n < 4)   bias = ba[n];
  else if (n < 8)   bias = bb[n - 4];
  else if (n < 72)  bias = bva[n - 8];
  else if (n < 136) bias = bvb[n - 72];
  else              bias = 0.f;
#pragma unroll
  for (int r = 0; r < 8; ++r) {
    int m = mBase + wave * 16 + r + 8 * hi;
    U[(size_t)m * N2PAD + n] = acc[r] + bias;
  }
}

// ============================================================================
// Memory update: factorized rank-1 + p5-normalize + mean over K.
// n_k = (S0_k * S1)^(1/5) factorization avoids materializing (B,4,256).
// One wave per batch row; lane j <-> u1 index j.
// U row layout: [alpha 0..3 | beta 4..7 | u0a 8..39 | u1a 40..71 | u0b 72..103 | u1b 104..135 | pad]
// ============================================================================
__device__ __forceinline__ float pow5abs(float x) {
  float a = fabsf(x);
  float t = a * a;
  return t * t * a;
}

__global__ __launch_bounds__(256) void nru_memupd(
    const float* __restrict__ U, const float* __restrict__ memin,
    float* __restrict__ memOut) {
  const int lane = threadIdx.x & 31;
  const int wave = threadIdx.x >> 5;
  const int b    = blockIdx.x * 8 + wave;
  const float* u = U + (size_t)b * N2PAD;

  float alphaL = (lane < 4) ? u[lane]     : 0.f;
  float betaL  = (lane < 4) ? u[4 + lane] : 0.f;
  float u0a = u[8 + lane];
  float u1a = u[40 + lane];
  float u0b = u[72 + lane];
  float u1b = u[104 + lane];

  // S1 = sum_j |u1|^5 over whole wave
  float s1a = pow5abs(u1a), s1b = pow5abs(u1b);
#pragma unroll
  for (int m = 1; m <= 16; m <<= 1) {
    s1a += __shfl_xor(s1a, m);
    s1b += __shfl_xor(s1b, m);
  }
  // S0_k = sum over 8-lane group of |u0|^5
  float s0a = pow5abs(u0a), s0b = pow5abs(u0b);
#pragma unroll
  for (int m = 1; m <= 4; m <<= 1) {
    s0a += __shfl_xor(s0a, m);
    s0b += __shfl_xor(s0b, m);
  }
  // n_k = (S0_k * S1)^(1/5); scale = coeff_k / max(n_k, eps)
  float na = powf(s0a * s1a, 0.2f);
  float nb = powf(s0b * s1b, 0.2f);
  int   k  = lane >> 3;
  float alpha_k = __shfl(alphaL, k);
  float beta_k  = __shfl(betaL,  k);
  float ca = (alpha_k / fmaxf(na, 1e-12f)) * u0a;   // lane = 8k+i
  float cb = (beta_k  / fmaxf(nb, 1e-12f)) * u0b;
  // pa[i] = sum_k ca(8k+i): reduce across groups (xor 8,16)
  ca += __shfl_xor(ca, 8);  ca += __shfl_xor(ca, 16);
  cb += __shfl_xor(cb, 8);  cb += __shfl_xor(cb, 16);

  // out[m] = mem[m] + 0.25*(pa[m/32]*u1a[m%32] - pb[m/32]*u1b[m%32])
#pragma unroll
  for (int t = 0; t < 8; ++t) {
    float pa = __shfl(ca, t);   // lane t holds pa[t]
    float pb = __shfl(cb, t);
    int m = t * 32 + lane;
    memOut[(size_t)b * MEMF + m] =
        memin[(size_t)b * MEMF + m] + 0.25f * (pa * u1a - pb * u1b);
  }
}

// ============================================================================
extern "C" void kernel_launch(void* const* d_in, const int* in_sizes, int n_in,
                              void* d_out, int out_size, void* d_ws, size_t ws_size,
                              hipStream_t stream) {
  const float* x    = (const float*)d_in[0];
  const float* h0   = (const float*)d_in[1];
  const float* mem  = (const float*)d_in[2];
  const float* Wh   = (const float*)d_in[3];
  const float* bh   = (const float*)d_in[4];
  const float* Wa   = (const float*)d_in[5];
  const float* ba   = (const float*)d_in[6];
  const float* Wb   = (const float*)d_in[7];
  const float* bb   = (const float*)d_in[8];
  const float* Wva  = (const float*)d_in[9];
  const float* bva  = (const float*)d_in[10];
  const float* Wvb  = (const float*)d_in[11];
  const float* bvb  = (const float*)d_in[12];

  float* memOut = (float*)d_out;                       // (B, 256)
  float* hOut   = memOut + (size_t)BATCH * MEMF;       // (B, 1024)
  float* U      = (float*)d_ws;                        // (B, 144) fp32 = 4.7 MB

  dim3 g1(HIDF / 128, BATCH / 128);  // (8, 64)
  nru_gemm1<<<g1, 256, 0, stream>>>(x, h0, mem, Wh, bh, hOut);

  dim3 g2(N2PAD / 16, BATCH / 128);  // (9, 64)
  nru_gemm2<<<g2, 256, 0, stream>>>(hOut, mem, Wa, Wb, Wva, Wvb,
                                    ba, bb, bva, bvb, U);

  nru_memupd<<<BATCH / 8, 256, 0, stream>>>(U, mem, memOut);
}
